// AdaAttention_67053029425581
// MI455X (gfx1250) — matrix operand order repaired and checked by reading, not compiled, not run on hardware
//
#include <hip/hip_runtime.h>

// ---------------------------------------------------------------------------
// CDNA5 (gfx1250) wave32 WMMA implementation of AdaAttention.
// Matrix math: v_wmma_f32_16x16x32_f16 (f16 in, f32 accum).
// Data movement: global_load_async_to_lds_b128 for pure f16 copies,
// double-buffered LDS staging for the main GEMM, batched f32->f16 A staging.
// ---------------------------------------------------------------------------

typedef __attribute__((ext_vector_type(16))) _Float16 v16h;
typedef __attribute__((ext_vector_type(8)))  _Float16 v8h;
typedef __attribute__((ext_vector_type(4)))  _Float16 v4h;
typedef __attribute__((ext_vector_type(8)))  float    v8f;
typedef __attribute__((ext_vector_type(4)))  int      v4i;

typedef __attribute__((address_space(1))) v4i glob_v4i;   // global int4
typedef __attribute__((address_space(3))) v4i lds_v4i;    // LDS int4

#if defined(__has_builtin)
#if __has_builtin(__builtin_amdgcn_global_load_async_to_lds_b128)
#define HAVE_ASYNC_LDS 1
#endif
#endif

__device__ __forceinline__ void copy16_g2lds(const _Float16* g, _Float16* lds) {
#ifdef HAVE_ASYNC_LDS
  __builtin_amdgcn_global_load_async_to_lds_b128(
      (glob_v4i*)(v4i*)g, (lds_v4i*)(v4i*)lds, 0, 0);
#else
  *(v8h*)lds = *(const v8h*)g;
#endif
}

__device__ __forceinline__ void wait_async_copies() {
#ifdef HAVE_ASYNC_LDS
#if __has_builtin(__builtin_amdgcn_s_wait_asynccnt)
  __builtin_amdgcn_s_wait_asynccnt(0);
#else
  asm volatile("s_wait_asynccnt 0x0" ::: "memory");
#endif
#endif
}

__device__ __forceinline__ v16h v16h_combine(v8h lo, v8h hi) {
  v16h r;
#pragma unroll
  for (int i = 0; i < 8; ++i) { r[i] = lo[i]; r[i + 8] = hi[i]; }
  return r;
}

__device__ __forceinline__ v8f wmma16x16x32(v16h a, v16h b, v8f c) {
  return __builtin_amdgcn_wmma_f32_16x16x32_f16(
      /*neg_a=*/false, a, /*neg_b=*/false, b,
      /*c_mod=*/(short)0, c, /*reuse_a=*/false, /*reuse_b=*/false);
}

// A-fragment per ISA table: lane holds row m=lane&15; K runs at (lane>>4)*8 and
// 16+(lane>>4)*8 (8 contiguous f16 each).  B-fragment: lane holds column
// n=lane&15; same two 8-wide K runs as A.  W stored [N,K] row-major, so
// both fragments are contiguous 16B runs -> ds_load_b128 / global_load_b128.

// ---------------------------------------------------------------------------
// fp32 -> f16 weight conversion
// ---------------------------------------------------------------------------
__global__ void cvt_f32_to_f16(const float* __restrict__ src,
                               _Float16* __restrict__ dst, int n) {
  int i = blockIdx.x * 256 + threadIdx.x;
  if (i < n) dst[i] = (_Float16)src[i];
}

// ---------------------------------------------------------------------------
// Generic GEMM: out = epilogue(A[M,K](f32) * W[N=512,K](f16)^T + bias)
// Block tile 128(M) x 64(N); 8 waves, wave w owns rows 16w..16w+15.
// Double-buffered K staging (64 at a time).  B goes through async-to-LDS
// (issued first, fire-and-forget); A is loaded in a batched two-phase
// gather (8 in-flight global_load_b128) then converted f32->f16 into LDS.
// Grid: x = N-block (8), y = M-block, so blocks sharing an A tile are
// launch-adjacent and A is served from L2.
// mode 0: relu -> f16 out; mode 1: identity -> f32 out; mode 2: tanh -> f32.
// Requires: M % 128 == 0, K % 64 == 0, N == 512.
// ---------------------------------------------------------------------------
__device__ __forceinline__ void stage_tiles(const float* __restrict__ A,
                                            const _Float16* __restrict__ W,
                                            int K, int mBase, int nBase, int kt,
                                            int tid, _Float16* Als,
                                            _Float16* Bls) {
  // B first: async copies are fire-and-forget, get them in flight earliest.
#pragma unroll
  for (int i = 0; i < 2; ++i) {
    const int c  = tid + 256 * i;      // 16B chunk id (512 total)
    const int n  = c >> 3;
    const int k8 = (c & 7) * 8;
    copy16_g2lds(&W[(size_t)(nBase + n) * K + kt + k8], &Bls[n * 64 + k8]);
  }
  // A: phase 1 — issue all 8 global loads into registers (batched latency).
  float4 tmp[8];
#pragma unroll
  for (int i = 0; i < 8; ++i) {
    const int c  = tid + 256 * i;      // float4 chunk id (2048 total)
    const int r  = c >> 4;             // row in tile
    const int c4 = (c & 15) * 4;       // K offset in tile
    tmp[i] = *(const float4*)&A[(size_t)(mBase + r) * K + kt + c4];
  }
  // A: phase 2 — convert and store to LDS.
#pragma unroll
  for (int i = 0; i < 8; ++i) {
    const int c  = tid + 256 * i;
    const int r  = c >> 4;
    const int c4 = (c & 15) * 4;
    _Float16* d = &Als[r * 64 + c4];
    d[0] = (_Float16)tmp[i].x; d[1] = (_Float16)tmp[i].y;
    d[2] = (_Float16)tmp[i].z; d[3] = (_Float16)tmp[i].w;
  }
}

__global__ __launch_bounds__(256)
void gemm_nt_512(const float* __restrict__ A, const _Float16* __restrict__ W,
                 const float* __restrict__ bias, int M, int K, int mode,
                 _Float16* __restrict__ out16, float* __restrict__ out32) {
  __shared__ __align__(16) _Float16 Als[2][128 * 64];
  __shared__ __align__(16) _Float16 Bls[2][64 * 64];

  const int tid  = threadIdx.x;
  const int wave = tid >> 5;
  const int lane = tid & 31;
  const int nBase = blockIdx.x * 64;
  const int mBase = blockIdx.y * 128;

  v8f acc[4] = {v8f{}, v8f{}, v8f{}, v8f{}};

  const int nstages = K >> 6;
  stage_tiles(A, W, K, mBase, nBase, 0, tid, Als[0], Bls[0]);

  for (int s = 0; s < nstages; ++s) {
    wait_async_copies();
    __syncthreads();
    if (s + 1 < nstages)
      stage_tiles(A, W, K, mBase, nBase, (s + 1) << 6, tid,
                  Als[(s + 1) & 1], Bls[(s + 1) & 1]);

    const _Float16* Ab = Als[s & 1];
    const _Float16* Bb = Bls[s & 1];
    const int m  = 16 * wave + (lane & 15);
    const int kh = lane >> 4;
#pragma unroll
    for (int ks = 0; ks < 64; ks += 32) {
      const v8h a0 = *(const v8h*)&Ab[m * 64 + ks + kh * 8];
      const v8h a1 = *(const v8h*)&Ab[m * 64 + ks + 16 + kh * 8];
      const v16h a = v16h_combine(a0, a1);
#pragma unroll
      for (int t = 0; t < 4; ++t) {
        const int nl = t * 16 + (lane & 15);
        const v8h b0 = *(const v8h*)&Bb[nl * 64 + ks + kh * 8];
        const v8h b1 = *(const v8h*)&Bb[nl * 64 + ks + 16 + kh * 8];
        acc[t] = wmma16x16x32(a, v16h_combine(b0, b1), acc[t]);
      }
    }
  }

  // epilogue: lane holds rows (lane>>4)*8+j, column lane&15 (per n-tile)
  const int colL = lane & 15;
  const int rl   = (lane >> 4) * 8;
#pragma unroll
  for (int t = 0; t < 4; ++t) {
    const int n  = nBase + t * 16 + colL;
    const float bn = bias[n];
#pragma unroll
    for (int j = 0; j < 8; ++j) {
      const int gr = mBase + wave * 16 + rl + j;
      const float v = acc[t][j] + bn;
      if (mode == 0)      out16[(size_t)gr * 512 + n] = (_Float16)fmaxf(v, 0.0f);
      else if (mode == 1) out32[(size_t)gr * 512 + n] = v;
      else                out32[(size_t)gr * 512 + n] = tanhf(v);
    }
  }
}

// ---------------------------------------------------------------------------
// Fused score kernel: for a 16-row slab, E = A * Wc^T (+bias +h_e),
// hA = tanh(E), logit = sum_a hA * W_al[a] + b_al.  N = K = 512.
// sent_mode 0: A = att16 slab, h_e row = b, writes logits[b, 1+s0+row]
// sent_mode 1: A = sentinel (f32) rows blk*16.., h_e row = global row,
//              writes logits[row, 0].
// ---------------------------------------------------------------------------
__global__ __launch_bounds__(256)
void score_kernel(const _Float16* __restrict__ A16, const float* __restrict__ A32,
                  const _Float16* __restrict__ Wn, const float* __restrict__ bias,
                  const float* __restrict__ h_e, const float* __restrict__ W_al,
                  const float* __restrict__ b_al, float* __restrict__ logits,
                  int sent_mode) {
  __shared__ __align__(16) _Float16 Als[16 * 512];
  __shared__ float logred[16];

  const int tid  = threadIdx.x;
  const int wave = tid >> 5;
  const int lane = tid & 31;

  int bIdx = 0, s0 = 0, aRow0;
  if (sent_mode == 0) {
    bIdx  = blockIdx.x / 6;
    s0    = (blockIdx.x % 6) * 16;
    aRow0 = bIdx * 96 + s0;
  } else {
    aRow0 = blockIdx.x * 16;
  }

  // stage the 16x512 A slab into LDS (f16)
  if (A16) {
#pragma unroll
    for (int i = 0; i < 4; ++i) {
      const int c  = tid + 256 * i;     // 16B chunk id (1024 total)
      const int r  = c >> 6;
      const int k8 = (c & 63) * 8;
      copy16_g2lds(&A16[(size_t)(aRow0 + r) * 512 + k8], &Als[r * 512 + k8]);
    }
    wait_async_copies();
  } else {
    float4 tmp[8];
#pragma unroll
    for (int i = 0; i < 8; ++i) {
      const int c  = tid + 256 * i;     // float4 chunk id (2048 total)
      const int r  = c >> 7;
      const int k4 = (c & 127) * 4;
      tmp[i] = *(const float4*)&A32[(size_t)(aRow0 + r) * 512 + k4];
    }
#pragma unroll
    for (int i = 0; i < 8; ++i) {
      const int c  = tid + 256 * i;
      const int r  = c >> 7;
      const int k4 = (c & 127) * 4;
      _Float16* d = &Als[r * 512 + k4];
      d[0] = (_Float16)tmp[i].x; d[1] = (_Float16)tmp[i].y;
      d[2] = (_Float16)tmp[i].z; d[3] = (_Float16)tmp[i].w;
    }
  }
  if (tid < 16) logred[tid] = 0.0f;
  __syncthreads();

  v8f acc[4] = {v8f{}, v8f{}, v8f{}, v8f{}};
  const int nb = wave * 64;
  const int m  = lane & 15;
  const int kh = lane >> 4;

  for (int ks = 0; ks < 512; ks += 32) {
    const v8h a0 = *(const v8h*)&Als[m * 512 + ks + kh * 8];
    const v8h a1 = *(const v8h*)&Als[m * 512 + ks + 16 + kh * 8];
    const v16h a = v16h_combine(a0, a1);
#pragma unroll
    for (int t = 0; t < 4; ++t) {
      const int n = nb + t * 16 + (lane & 15);
      const v8h* wp = (const v8h*)&Wn[(size_t)n * 512 + ks + kh * 8];
      acc[t] = wmma16x16x32(a, v16h_combine(wp[0], wp[2]), acc[t]);
    }
  }

  // epilogue: +bias +h_e, tanh, weight by W_al, reduce over columns
  float partial[8] = {0, 0, 0, 0, 0, 0, 0, 0};
  const int colL = lane & 15;
  const int rl   = (lane >> 4) * 8;
#pragma unroll
  for (int t = 0; t < 4; ++t) {
    const int n = nb + t * 16 + colL;
    const float bn  = bias[n];
    const float wal = W_al[n];
#pragma unroll
    for (int j = 0; j < 8; ++j) {
      const int row   = rl + j;
      const int heRow = sent_mode ? (aRow0 + row) : bIdx;
      const float v = acc[t][j] + bn + h_e[(size_t)heRow * 512 + n];
      partial[j] += tanhf(v) * wal;
    }
  }
  // butterfly reduce over the 16 lanes of each half-wave (columns)
#pragma unroll
  for (int msk = 1; msk < 16; msk <<= 1) {
#pragma unroll
    for (int j = 0; j < 8; ++j)
      partial[j] += __shfl_xor(partial[j], msk, 32);
  }
  if (colL == 0) {
#pragma unroll
    for (int j = 0; j < 8; ++j) atomicAdd(&logred[rl + j], partial[j]);
  }
  __syncthreads();
  if (tid < 16) {
    const float lv = logred[tid] + b_al[0];
    const int idx = sent_mode ? ((aRow0 + tid) * 97)
                              : (bIdx * 97 + 1 + s0 + tid);
    logits[idx] = lv;
  }
}

// ---------------------------------------------------------------------------
// Softmax over 97 slots + context sum:
// attenIn = alpha0*sentinel + sum_s alpha[s+1]*att + h   (vectorized over R)
// ---------------------------------------------------------------------------
__global__ __launch_bounds__(128)
void softmax_ctx_kernel(const float* __restrict__ logits,
                        const float* __restrict__ sentinel,
                        const _Float16* __restrict__ att16,
                        const float* __restrict__ h,
                        float* __restrict__ attenIn) {
  __shared__ float sl[97];
  __shared__ float alpha[97];
  const int b = blockIdx.x;
  const int tid = threadIdx.x;
  if (tid < 97) sl[tid] = logits[b * 97 + tid];
  __syncthreads();
  float mx = -3.0e38f;
  for (int t = 0; t < 97; ++t) mx = fmaxf(mx, sl[t]);
  float sum = 0.0f;
  for (int t = 0; t < 97; ++t) sum += expf(sl[t] - mx);
  if (tid < 97) alpha[tid] = expf(sl[tid] - mx) / sum;
  __syncthreads();

  const int r4 = tid * 4;                 // 128 threads x 4 cols = 512
  const float4 sv = *(const float4*)&sentinel[(size_t)b * 512 + r4];
  const float4 hv = *(const float4*)&h[(size_t)b * 512 + r4];
  const float a0 = alpha[0];
  float ac0 = a0 * sv.x, ac1 = a0 * sv.y, ac2 = a0 * sv.z, ac3 = a0 * sv.w;
  for (int s = 0; s < 96; ++s) {
    const v4h av = *(const v4h*)&att16[(size_t)(b * 96 + s) * 512 + r4];
    const float as = alpha[s + 1];
    ac0 += as * (float)av[0];
    ac1 += as * (float)av[1];
    ac2 += as * (float)av[2];
    ac3 += as * (float)av[3];
  }
  float4 outv = make_float4(ac0 + hv.x, ac1 + hv.y, ac2 + hv.z, ac3 + hv.w);
  *(float4*)&attenIn[(size_t)b * 512 + r4] = outv;
}

// ---------------------------------------------------------------------------
// Host launch
// ---------------------------------------------------------------------------
extern "C" void kernel_launch(void* const* d_in, const int* in_sizes, int n_in,
                              void* d_out, int out_size, void* d_ws, size_t ws_size,
                              hipStream_t stream) {
  (void)in_sizes; (void)n_in; (void)out_size; (void)ws_size;
  const float* h         = (const float*)d_in[0];
  const float* sentinel  = (const float*)d_in[1];
  const float* att_feats = (const float*)d_in[2];
  const float* W_ae = (const float*)d_in[3];  const float* b_ae = (const float*)d_in[4];
  const float* W_c  = (const float*)d_in[5];  const float* b_c  = (const float*)d_in[6];
  const float* W_s  = (const float*)d_in[7];  const float* b_s  = (const float*)d_in[8];
  const float* W_h  = (const float*)d_in[9];  const float* b_h  = (const float*)d_in[10];
  const float* W_al = (const float*)d_in[11]; const float* b_al = (const float*)d_in[12];
  const float* W_o  = (const float*)d_in[13]; const float* b_o  = (const float*)d_in[14];

  const int B = 1024, S = 96, F = 2048, R = 512;

  // workspace carve
  char* p = (char*)d_ws;
  auto carve = [&](size_t bytes) {
    void* r = (void*)p;
    p += (bytes + 255) & ~(size_t)255;
    return r;
  };
  _Float16* W_ae16 = (_Float16*)carve((size_t)R * F * 2);
  _Float16* W_c16  = (_Float16*)carve((size_t)R * R * 2);
  _Float16* W_s16  = (_Float16*)carve((size_t)R * R * 2);
  _Float16* W_h16  = (_Float16*)carve((size_t)R * R * 2);
  _Float16* W_o16  = (_Float16*)carve((size_t)R * R * 2);
  _Float16* att16  = (_Float16*)carve((size_t)B * S * R * 2);
  float*    h_e    = (float*)carve((size_t)B * R * 4);
  float*    logits = (float*)carve((size_t)B * 97 * 4);
  float*    attenIn= (float*)carve((size_t)B * R * 4);

  // 1) weight conversion to f16
  cvt_f32_to_f16<<<(R * F + 255) / 256, 256, 0, stream>>>(W_ae, W_ae16, R * F);
  cvt_f32_to_f16<<<(R * R + 255) / 256, 256, 0, stream>>>(W_c, W_c16, R * R);
  cvt_f32_to_f16<<<(R * R + 255) / 256, 256, 0, stream>>>(W_s, W_s16, R * R);
  cvt_f32_to_f16<<<(R * R + 255) / 256, 256, 0, stream>>>(W_h, W_h16, R * R);
  cvt_f32_to_f16<<<(R * R + 255) / 256, 256, 0, stream>>>(W_o, W_o16, R * R);

  // 2) h_e = h @ W_h^T + b_h            [1024,512]  (f32 out)
  gemm_nt_512<<<dim3(8, B / 128), 256, 0, stream>>>(
      h, W_h16, b_h, B, R, /*mode=*/1, nullptr, h_e);

  // 3) att = relu(att_feats @ W_ae^T + b_ae)   [98304,512] -> f16
  gemm_nt_512<<<dim3(8, (B * S) / 128), 256, 0, stream>>>(
      att_feats, W_ae16, b_ae, B * S, F, /*mode=*/0, att16, nullptr);

  // 4) sentinel slot logits[:,0]
  score_kernel<<<B / 16, 256, 0, stream>>>(
      nullptr, sentinel, W_s16, b_s, h_e, W_al, b_al, logits, /*sent_mode=*/1);

  // 5) attention slots logits[:,1..96] (fused ctx2att GEMM + tanh + alpha_net)
  score_kernel<<<B * (S / 16), 256, 0, stream>>>(
      att16, nullptr, W_c16, b_c, h_e, W_al, b_al, logits, /*sent_mode=*/0);

  // 6) softmax + weighted context + residual h
  softmax_ctx_kernel<<<B, 128, 0, stream>>>(logits, sentinel, att16, h, attenIn);

  // 7) out = tanh(attenIn @ W_o^T + b_o)   (f32 out -> d_out)
  gemm_nt_512<<<dim3(8, B / 128), 256, 0, stream>>>(
      attenIn, W_o16, b_o, B, R, /*mode=*/2, nullptr, (float*)d_out);
}
